// GRUHead_52664888984242
// MI455X (gfx1250) — matrix operand order repaired and checked by reading, not compiled, not run on hardware
//
#include <hip/hip_runtime.h>

// GRU fused kernel for MI455X (gfx1250, wave32, WMMA).
// T=128, B=4096, H=64, gates G=3H=192 in order r,z,n.
#define T_STEPS 128
#define BATCH   4096
#define HID     64

typedef __attribute__((ext_vector_type(16))) __bf16 v16bf;
typedef __attribute__((ext_vector_type(8)))  __bf16 v8bf;
typedef __attribute__((ext_vector_type(8)))  float  v8f;

#define ROWS_PER_WAVE 16
#define WAVES          4
#define BLOCK         (WAVES * 32)
#define ROWS_PER_BLOCK (ROWS_PER_WAVE * WAVES)
#define GRID          (BATCH / ROWS_PER_BLOCK)   // 64 blocks

__device__ __forceinline__ float fast_sigmoid(float x) {
    return 1.0f / (1.0f + __expf(-x));
}
__device__ __forceinline__ float fast_tanh(float x) {
    float a = fabsf(x);
    float e = __expf(-2.0f * a);
    float r = (1.0f - e) / (1.0f + e);
    return copysignf(r, x);
}

// Build a 16-bit A-fragment lane vector from 16 fp32 values:
// v[0..7] = floats a,b ; v[8..15] = floats c,d
__device__ __forceinline__ v16bf cvt16(float4 a, float4 b, float4 c, float4 d) {
    v16bf v;
    v[0]  = (__bf16)a.x; v[1]  = (__bf16)a.y; v[2]  = (__bf16)a.z; v[3]  = (__bf16)a.w;
    v[4]  = (__bf16)b.x; v[5]  = (__bf16)b.y; v[6]  = (__bf16)b.z; v[7]  = (__bf16)b.w;
    v[8]  = (__bf16)c.x; v[9]  = (__bf16)c.y; v[10] = (__bf16)c.z; v[11] = (__bf16)c.w;
    v[12] = (__bf16)d.x; v[13] = (__bf16)d.y; v[14] = (__bf16)d.z; v[15] = (__bf16)d.w;
    return v;
}

#define WMMA_BF16(A, B, C) \
    __builtin_amdgcn_wmma_f32_16x16x32_bf16(false, (A), false, (B), (short)0, (C), false, false)

__global__ __launch_bounds__(BLOCK) void gru_fused(
    const float* __restrict__ Hseq,  // [T,B,H]
    const float* __restrict__ W_ih,  // [3H,H]
    const float* __restrict__ W_hh,  // [3H,H]
    const float* __restrict__ b_ih,  // [3H]
    const float* __restrict__ b_hh,  // [3H]
    const float* __restrict__ W_out, // [1,H]
    const float* __restrict__ b_out, // [1]
    float* __restrict__ out)         // [B]
{
    // Weight B-fragments: 2 matrices x 12 N-tiles x 2 K-halves = 48 frags,
    // each 32 lanes x 16 bf16 (32B/lane) = 1KB -> 48KB.
    // (The compiler hoists these into the extended 1024-VGPR file for the T loop.)
    __shared__ v16bf wfrag[48 * 32];
    __shared__ float brz[128];           // b_ih+b_hh for r,z gates
    __shared__ float bxn[64];            // b_ih for n gate
    __shared__ float bhn[64];            // b_hh for n gate
    __shared__ v8bf  hbuf8[WAVES][16 * 8];  // h state, bf16 [16 rows][64 cols], per wave

    const int tid  = threadIdx.x;
    const int lane = tid & 31;
    const int wave = tid >> 5;
    const int arow  = lane & 15;   // A-fragment row / C-layout column-in-tile
    const int ahalf = lane >> 4;   // lane half selects K interleave

    // ---- Prologue: stage weights as bf16 WMMA B-fragments in LDS ----
    // B layout (K x N, 16-bit): lanes 0-15 hold K=0..7,16..23 of column N=lane,
    // lanes 16-31 hold K=8..15,24..31 of column N=lane-16. Column g of W^T is W[g][0..63]
    // (contiguous), so each lane converts two runs of 8 fp32.
    for (int e = tid; e < 48 * 32; e += BLOCK) {
        int f = e >> 5, l = e & 31;
        int m = f / 24, rem = f % 24;      // m: 0=W_ih, 1=W_hh
        int j = rem >> 1, k = rem & 1;     // j: N-tile 0..11, k: K-half
        int g  = 16 * j + (l & 15);
        int kk = 32 * k + ((l >> 4) << 3);
        const float* src = (m ? W_hh : W_ih) + g * HID + kk;
        v16bf v;
#pragma unroll
        for (int i = 0; i < 16; ++i) v[i] = (__bf16)src[i + (i >= 8 ? 8 : 0)];
        wfrag[e] = v;
    }
    if (tid < 128) brz[tid] = b_ih[tid] + b_hh[tid];
    if (tid < 64)  { bxn[tid] = b_ih[128 + tid]; bhn[tid] = b_hh[128 + tid]; }
    // zero h state (h0 = 0): 128 v8bf per wave / 32 lanes = 4 each
#pragma unroll
    for (int i = 0; i < 4; ++i) hbuf8[wave][lane * 4 + i] = (v8bf)0;
    __syncthreads();

    const int rb = blockIdx.x * ROWS_PER_BLOCK + wave * ROWS_PER_WAVE;
    // This lane's Hseq row base (A-fragment: lane l and l+16 share row l&15)
    const float4* xrow = (const float4*)(Hseq + (size_t)(rb + arow) * HID);
    __bf16* hb = (__bf16*)&hbuf8[wave][0];
    const int c0 = 2 * ahalf;            // float4 chunk of 8*ahalf floats
    const size_t tstride = (size_t)(BATCH * HID / 4);   // float4 per timestep

    v8f hC[4], rT[4], zT[4];
#pragma unroll
    for (int j = 0; j < 4; ++j)
#pragma unroll
        for (int e = 0; e < 8; ++e) hC[j][e] = 0.0f;

    // ---- software pipeline: preload X raw data for t=0 ----
    float4 xf[8];
    {
        const float4* xp0 = xrow;
        xf[0] = xp0[c0];      xf[1] = xp0[c0 + 1];
        xf[2] = xp0[c0 + 4];  xf[3] = xp0[c0 + 5];
        xf[4] = xp0[c0 + 8];  xf[5] = xp0[c0 + 9];
        xf[6] = xp0[c0 + 12]; xf[7] = xp0[c0 + 13];
    }

#pragma unroll 1
    for (int t = 0; t < T_STEPS; ++t) {
        // Issue next step's global loads first: they retire behind this
        // step's 48 WMMAs (full-step latency cover at 1 wave/SIMD).
        const int tn = (t + 1 < T_STEPS) ? (t + 1) : t;
        const float4* xpn = xrow + (size_t)tn * tstride;
        float4 xn0 = xpn[c0];      float4 xn1 = xpn[c0 + 1];
        float4 xn2 = xpn[c0 + 4];  float4 xn3 = xpn[c0 + 5];
        float4 xn4 = xpn[c0 + 8];  float4 xn5 = xpn[c0 + 9];
        float4 xn6 = xpn[c0 + 12]; float4 xn7 = xpn[c0 + 13];
        // warm L2 for t+2
        __builtin_prefetch((const void*)(xpn + tstride), 0, 1);

        // A-fragments of X tile (16x64 fp32 -> 2 bf16 K-fragments) from the
        // registers loaded last iteration.
        v16bf xa0 = cvt16(xf[0], xf[1], xf[2], xf[3]);
        v16bf xa1 = cvt16(xf[4], xf[5], xf[6], xf[7]);

        // A-fragments of h (bf16 straight from LDS; chunk = 8 bf16 = 16B)
        v16bf ha0, ha1;
        {
            v8bf lo0 = hbuf8[wave][arow * 8 + ahalf];
            v8bf hi0 = hbuf8[wave][arow * 8 + ahalf + 2];
            v8bf lo1 = hbuf8[wave][arow * 8 + 4 + ahalf];
            v8bf hi1 = hbuf8[wave][arow * 8 + 4 + ahalf + 2];
#pragma unroll
            for (int i = 0; i < 8; ++i) {
                ha0[i] = lo0[i]; ha0[i + 8] = hi0[i];
                ha1[i] = lo1[i]; ha1[i + 8] = hi1[i];
            }
        }

        // ---- r,z gates: merged bias + X*W_ih^T + h*W_hh^T, then sigmoid ----
#pragma unroll
        for (int j = 0; j < 8; ++j) {
            float bias = brz[16 * j + arow];
            v8f acc;
#pragma unroll
            for (int e = 0; e < 8; ++e) acc[e] = bias;
            acc = WMMA_BF16(xa0, wfrag[(j * 2 + 0) * 32 + lane], acc);
            acc = WMMA_BF16(xa1, wfrag[(j * 2 + 1) * 32 + lane], acc);
            acc = WMMA_BF16(ha0, wfrag[(24 + j * 2 + 0) * 32 + lane], acc);
            acc = WMMA_BF16(ha1, wfrag[(24 + j * 2 + 1) * 32 + lane], acc);
            v8f s;
#pragma unroll
            for (int e = 0; e < 8; ++e) s[e] = fast_sigmoid(acc[e]);
            if (j < 4) rT[j] = s; else zT[j - 4] = s;
        }

        // ---- n gate needs xn and hn separate: n = tanh(xn + r*hn) ----
#pragma unroll
        for (int j = 0; j < 4; ++j) {
            int jj = 8 + j;
            float bx = bxn[16 * j + arow];
            float bh = bhn[16 * j + arow];
            v8f ax, ah;
#pragma unroll
            for (int e = 0; e < 8; ++e) { ax[e] = bx; ah[e] = bh; }
            ax = WMMA_BF16(xa0, wfrag[(jj * 2 + 0) * 32 + lane], ax);
            ax = WMMA_BF16(xa1, wfrag[(jj * 2 + 1) * 32 + lane], ax);
            ah = WMMA_BF16(ha0, wfrag[(24 + jj * 2 + 0) * 32 + lane], ah);
            ah = WMMA_BF16(ha1, wfrag[(24 + jj * 2 + 1) * 32 + lane], ah);
#pragma unroll
            for (int e = 0; e < 8; ++e) {
                float nn = fast_tanh(ax[e] + rT[j][e] * ah[e]);
                float z  = zT[j][e];
                hC[j][e] = (1.0f - z) * nn + z * hC[j][e];
            }
        }

        // ---- write h back to per-wave LDS (C layout -> row-major bf16) ----
        // C layout: VGPR e, lane l -> row = e + 8*(l>>4), col = 16j + (l&15)
#pragma unroll
        for (int j = 0; j < 4; ++j)
#pragma unroll
            for (int e = 0; e < 8; ++e) {
                int row = e + 8 * ahalf;
                int col = 16 * j + arow;
                hb[row * HID + col] = (__bf16)hC[j][e];
            }
        // same-wave LDS ops are in-order; next iteration's ha loads see new h

        // rotate the software pipeline
        xf[0] = xn0; xf[1] = xn1; xf[2] = xn2; xf[3] = xn3;
        xf[4] = xn4; xf[5] = xn5; xf[6] = xn6; xf[7] = xn7;
    }

    // ---- output head: out[b] = h_last[b,:] . W_out + b_out ----
    if (lane < 16) {
        float acc = b_out[0];
        const __bf16* hp = hb + lane * HID;
#pragma unroll
        for (int c = 0; c < HID; ++c) acc += (float)hp[c] * W_out[c];
        out[rb + lane] = acc;
    }
}

extern "C" void kernel_launch(void* const* d_in, const int* in_sizes, int n_in,
                              void* d_out, int out_size, void* d_ws, size_t ws_size,
                              hipStream_t stream) {
    const float* Hseq  = (const float*)d_in[0];
    const float* W_ih  = (const float*)d_in[1];
    const float* W_hh  = (const float*)d_in[2];
    const float* b_ih  = (const float*)d_in[3];
    const float* b_hh  = (const float*)d_in[4];
    const float* W_out = (const float*)d_in[5];
    const float* b_out = (const float*)d_in[6];
    float* out = (float*)d_out;
    (void)in_sizes; (void)n_in; (void)out_size; (void)d_ws; (void)ws_size;

    gru_fused<<<GRID, BLOCK, 0, stream>>>(Hseq, W_ih, W_hh, b_ih, b_hh,
                                          W_out, b_out, out);
}
